// FraSICL_42322607735332
// MI455X (gfx1250) — matrix operand order repaired and checked by reading, not compiled, not run on hardware
//
#include <hip/hip_runtime.h>
#include <hip/hip_bf16.h>
#include <math.h>

typedef __bf16 v16bf __attribute__((ext_vector_type(16)));
typedef __bf16 v8bf  __attribute__((ext_vector_type(8)));
typedef __bf16 v4bf  __attribute__((ext_vector_type(4)));
typedef float  v8f   __attribute__((ext_vector_type(8)));

#define B_MOL   512
#define F_DIM   256
#define PH_DIM  512
#define H_DIM   256
#define FFN_DIM 1024
#define MAXP    24
#define P_PAIRS 6336

__device__ __forceinline__ v4bf pack_bf16x4(float4 f)
{
    v4bf r;
    r[0] = (__bf16)f.x; r[1] = (__bf16)f.y; r[2] = (__bf16)f.z; r[3] = (__bf16)f.w;
    return r;
}

// ---------------------------------------------------------------------------
// C[M,N] = epilogue( A[M,K] * B )  via v_wmma_f32_16x16x32_bf16
//   transB==0: B is [K,N] row-major (weights);  transB==1: B is [N,K] (C=A*B^T)
//   epilogue: (+bias[n]) (+resid[m,n]) (act==2 -> tanh-GELU)
// Requirements (all call sites satisfy them): K % 32 == 0, N % 64 == 0.
// Block: 256 threads (8 waves), tile 128x64x32, bf16 staged through LDS,
// register double-buffered global->LDS staging (branch-free, vectorized).
// ---------------------------------------------------------------------------
__global__ __launch_bounds__(256)
void gemm_wmma_bf16(const float* __restrict__ A, const float* __restrict__ Bm,
                    float* __restrict__ C, const float* __restrict__ bias,
                    const float* __restrict__ resid,
                    int M, int N, int K, int transB, int act)
{
    const int LDT = 40;                       // padded row stride (bf16), 80B rows: 16B aligned
    __shared__ __align__(16) __bf16 sA[128 * 40];
    __shared__ __align__(16) __bf16 sB[64 * 40];

    const int tid  = threadIdx.x;
    const int w    = tid >> 5;
    const int lane = tid & 31;
    const int m0   = blockIdx.y * 128;
    const int n0   = blockIdx.x * 64;

    v8f acc[4];
    #pragma unroll
    for (int t = 0; t < 4; ++t)
        #pragma unroll
        for (int i = 0; i < 8; ++i) acc[t][i] = 0.0f;

    // staging registers (double buffer)
    float4 ra[4];
    float  rb[8];

    // A mapping: idx = i*256+tid -> m = idx>>3 (0..127), kq4 = (idx&7)*4
    const int a_m   = tid >> 3;
    const int a_k4  = (tid & 7) * 4;
    // B transB mapping: idx = i*256+tid -> n = idx>>3 (0..63), kq4 = (idx&7)*4
    // B normal mapping: idx = i*256+tid -> n = idx&63, k4 = (idx>>6)*4
    const int bn_n  = tid & 63;
    const int bn_k4 = (tid >> 6) * 4;

    auto load_tiles = [&](int kk) {
        #pragma unroll
        for (int i = 0; i < 4; ++i) {
            int gm = m0 + a_m + i * 32;           // rows advance by 32 per iter (i*256 idx)
            gm = gm < M - 1 ? gm : M - 1;
            ra[i] = *(const float4*)(A + (size_t)gm * K + kk + a_k4);
        }
        if (transB) {
            #pragma unroll
            for (int i = 0; i < 2; ++i) {
                int gn = n0 + (tid >> 3) + i * 32;
                gn = gn < N - 1 ? gn : N - 1;
                float4 f = *(const float4*)(Bm + (size_t)gn * K + kk + a_k4);
                rb[i * 4 + 0] = f.x; rb[i * 4 + 1] = f.y; rb[i * 4 + 2] = f.z; rb[i * 4 + 3] = f.w;
            }
        } else {
            #pragma unroll
            for (int i = 0; i < 2; ++i) {
                int k4 = bn_k4 + i * 16;          // i=0: k 0..12, i=1: k 16..28
                const float* src = Bm + (size_t)(kk + k4) * N + n0 + bn_n;
                rb[i * 4 + 0] = src[0];
                rb[i * 4 + 1] = src[(size_t)N];
                rb[i * 4 + 2] = src[(size_t)2 * N];
                rb[i * 4 + 3] = src[(size_t)3 * N];
            }
        }
    };

    auto store_tiles = [&]() {
        #pragma unroll
        for (int i = 0; i < 4; ++i) {
            int m = a_m + i * 32;
            *(v4bf*)(sA + m * LDT + a_k4) = pack_bf16x4(ra[i]);     // 8B-aligned ds_store_b64
        }
        if (transB) {
            #pragma unroll
            for (int i = 0; i < 2; ++i) {
                int n = (tid >> 3) + i * 32;
                v4bf p; p[0] = (__bf16)rb[i*4+0]; p[1] = (__bf16)rb[i*4+1];
                p[2] = (__bf16)rb[i*4+2]; p[3] = (__bf16)rb[i*4+3];
                *(v4bf*)(sB + n * LDT + a_k4) = p;
            }
        } else {
            #pragma unroll
            for (int i = 0; i < 2; ++i) {
                int k4 = bn_k4 + i * 16;
                v4bf p; p[0] = (__bf16)rb[i*4+0]; p[1] = (__bf16)rb[i*4+1];
                p[2] = (__bf16)rb[i*4+2]; p[3] = (__bf16)rb[i*4+3];
                *(v4bf*)(sB + bn_n * LDT + k4) = p;
            }
        }
    };

    load_tiles(0);

    for (int kk = 0; kk < K; kk += 32) {
        store_tiles();
        __syncthreads();

        // prefetch next K-step while this one computes (loads overlap ds+wmma)
        if (kk + 32 < K) load_tiles(kk + 32);

        const int half = lane >> 4;     // lane half selects K sub-range per ISA layout
        const int l16  = lane & 15;

        // A fragment: row = w*16 + l16 ; e<8 -> k=half*8+e ; e>=8 -> k=16+half*8+(e-8)
        const v8bf* pa = (const v8bf*)(sA + (w * 16 + l16) * LDT);
        v8bf alo = pa[half];            // K = half*8 .. +7
        v8bf ahi = pa[2 + half];        // K = 16 + half*8 .. +7
        v16bf af;
        #pragma unroll
        for (int i = 0; i < 8; ++i) { af[i] = alo[i]; af[8 + i] = ahi[i]; }

        #pragma unroll
        for (int t = 0; t < 4; ++t) {
            // B fragment: col = t*16 + l16 ; e<8 -> k=half*8+e ; e>=8 -> k=16+half*8+(e-8)
            const v8bf* pb = (const v8bf*)(sB + (t * 16 + l16) * LDT);
            v8bf blo = pb[half];
            v8bf bhi = pb[2 + half];
            v16bf bf;
            #pragma unroll
            for (int i = 0; i < 8; ++i) { bf[i] = blo[i]; bf[8 + i] = bhi[i]; }
            acc[t] = __builtin_amdgcn_wmma_f32_16x16x32_bf16(
                         false, af, false, bf, (short)0, acc[t], false, false);
        }
        __syncthreads();
    }

    // ---- epilogue: C/D layout VGPR r -> M = r + 8*(lane>=16), N = lane%16
    const int half = lane >> 4;
    const int l16  = lane & 15;
    #pragma unroll
    for (int r = 0; r < 8; ++r) {
        int gm = m0 + w * 16 + r + half * 8;
        #pragma unroll
        for (int t = 0; t < 4; ++t) {
            int gn = n0 + t * 16 + l16;
            if (gm < M && gn < N) {
                float v = acc[t][r];
                if (bias)  v += bias[gn];
                if (resid) v += resid[(size_t)gm * N + gn];
                if (act == 2) {
                    float u = v;
                    v = 0.5f * u * (1.0f + tanhf(0.7978845608028654f * (u + 0.044715f * u * u * u)));
                }
                C[(size_t)gm * N + gn] = v;
            }
        }
    }
}

// ---------------------------------------------------------------------------
// pairs[p,:] = frag_raw[2p,:] + frag_raw[2p+1,:]
__global__ void pairs_sum(const float* __restrict__ fr, float* __restrict__ pairs)
{
    int p = blockIdx.x, f = threadIdx.x;
    pairs[(size_t)p * F_DIM + f] =
        fr[(size_t)(2 * p) * F_DIM + f] + fr[(size_t)(2 * p + 1) * F_DIM + f];
}

// BatchNorm batch stats (train mode, biased var, eps=1e-6). 32 cols per block.
__global__ __launch_bounds__(256)
void bn_stats(const float* __restrict__ Hm, float* __restrict__ mu,
              float* __restrict__ inv, int M, int N)
{
    __shared__ float s1[256], s2[256];
    int cl = threadIdx.x & 31;
    int rl = threadIdx.x >> 5;
    int col = blockIdx.x * 32 + cl;
    float a = 0.f, b = 0.f;
    for (int r = rl; r < M; r += 8) {
        float v = Hm[(size_t)r * N + col];
        a += v; b += v * v;
    }
    s1[threadIdx.x] = a; s2[threadIdx.x] = b;
    __syncthreads();
    if (threadIdx.x < 32) {
        float aa = 0.f, bb = 0.f;
        for (int i = 0; i < 8; ++i) { aa += s1[i * 32 + cl]; bb += s2[i * 32 + cl]; }
        float m = aa / (float)M;
        float var = bb / (float)M - m * m;
        mu[col] = m;
        inv[col] = rsqrtf(var + 1e-6f);
    }
}

__global__ void bn_apply_relu(float* __restrict__ Hm, const float* __restrict__ mu,
                              const float* __restrict__ inv, const float* __restrict__ g,
                              const float* __restrict__ be, int M, int N)
{
    size_t i = (size_t)blockIdx.x * blockDim.x + threadIdx.x;
    if (i >= (size_t)M * N) return;
    int col = (int)(i % N);
    float v = (Hm[i] - mu[col]) * inv[col] * g[col] + be[col];
    Hm[i] = v > 0.f ? v : 0.f;
}

// row-wise x / max(||x||2, 1e-12), N == 256
__global__ __launch_bounds__(256)
void l2norm_rows(const float* __restrict__ in, float* __restrict__ out)
{
    __shared__ float red[256];
    int row = blockIdx.x, t = threadIdx.x;
    float v = in[(size_t)row * 256 + t];
    red[t] = v * v;
    __syncthreads();
    for (int s = 128; s > 0; s >>= 1) { if (t < s) red[t] += red[t + s]; __syncthreads(); }
    float sc = 1.0f / fmaxf(sqrtf(red[0]), 1e-12f);
    out[(size_t)row * 256 + t] = v * sc;
}

// LayerNorm rows, H == 256, eps=1e-5
__global__ __launch_bounds__(256)
void layernorm_rows(const float* __restrict__ in, float* __restrict__ out,
                    const float* __restrict__ g, const float* __restrict__ b)
{
    __shared__ float red[256];
    int row = blockIdx.x, t = threadIdx.x;
    float v = in[(size_t)row * 256 + t];
    red[t] = v;
    __syncthreads();
    for (int s = 128; s > 0; s >>= 1) { if (t < s) red[t] += red[t + s]; __syncthreads(); }
    float mu = red[0] * (1.0f / 256.0f);
    __syncthreads();
    float d = v - mu;
    red[t] = d * d;
    __syncthreads();
    for (int s = 128; s > 0; s >>= 1) { if (t < s) red[t] += red[t + s]; __syncthreads(); }
    float var = red[0] * (1.0f / 256.0f);
    out[(size_t)row * 256 + t] = d * rsqrtf(var + 1e-5f) * g[t] + b[t];
}

__global__ void zero_f32(float* __restrict__ p, size_t n)
{
    size_t i = (size_t)blockIdx.x * blockDim.x + threadIdx.x;
    if (i < n) p[i] = 0.0f;
}

__global__ void scatter_pairs(const float* __restrict__ pairs, const int* __restrict__ pm,
                              const int* __restrict__ pp, float* __restrict__ batched)
{
    int p = blockIdx.x, f = threadIdx.x;
    int row = pm[p] * MAXP + pp[p];
    batched[(size_t)row * F_DIM + f] = pairs[(size_t)p * F_DIM + f];
}

// one block per (molecule b, head h): seq=24, dh=32, no mask (matches reference)
__global__ __launch_bounds__(256)
void attention(const float* __restrict__ q, const float* __restrict__ k,
               const float* __restrict__ v, float* __restrict__ att)
{
    __shared__ float qs[24][32], ks[24][32], vs[24][32], sc[24][24];
    int b = blockIdx.x >> 3;
    int h = blockIdx.x & 7;
    int t = threadIdx.x;
    for (int idx = t; idx < 24 * 32; idx += 256) {
        int s_ = idx >> 5, d = idx & 31;
        size_t off = (size_t)(b * 24 + s_) * H_DIM + h * 32 + d;
        qs[s_][d] = q[off]; ks[s_][d] = k[off]; vs[s_][d] = v[off];
    }
    __syncthreads();
    for (int idx = t; idx < 24 * 24; idx += 256) {
        int i = idx / 24, j = idx % 24;
        float s = 0.f;
        #pragma unroll
        for (int d = 0; d < 32; ++d) s += qs[i][d] * ks[j][d];
        sc[i][j] = s * 0.17677669529663687f;   // 1/sqrt(32)
    }
    __syncthreads();
    if (t < 24) {
        float mx = -1e30f;
        for (int j = 0; j < 24; ++j) mx = fmaxf(mx, sc[t][j]);
        float sum = 0.f;
        for (int j = 0; j < 24; ++j) { float e = expf(sc[t][j] - mx); sc[t][j] = e; sum += e; }
        float is = 1.0f / sum;
        for (int j = 0; j < 24; ++j) sc[t][j] *= is;
    }
    __syncthreads();
    for (int idx = t; idx < 24 * 32; idx += 256) {
        int s_ = idx >> 5, d = idx & 31;
        float o = 0.f;
        #pragma unroll
        for (int j = 0; j < 24; ++j) o += sc[s_][j] * vs[j][d];
        att[(size_t)(b * 24 + s_) * H_DIM + h * 32 + d] = o;
    }
}

// masked mean over valid positions
__global__ __launch_bounds__(256)
void readout_mean(const float* __restrict__ enc, const int* __restrict__ sbn,
                  float* __restrict__ out)
{
    int b = blockIdx.x, f = threadIdx.x;
    int c = sbn[b]; if (c < 1) c = 1;
    float s = 0.f;
    for (int p = 0; p < c; ++p) s += enc[(size_t)(b * MAXP + p) * F_DIM + f];
    out[(size_t)b * F_DIM + f] = s / (float)c;
}

// ---------------------------------------------------------------------------
static inline dim3 gemm_grid(int M, int N) { return dim3((N + 63) / 64, (M + 127) / 128); }

extern "C" void kernel_launch(void* const* d_in, const int* in_sizes, int n_in,
                              void* d_out, int out_size, void* d_ws, size_t ws_size,
                              hipStream_t stream)
{
    (void)in_sizes; (void)n_in; (void)out_size; (void)ws_size;
    const float* mol_emb  = (const float*)d_in[0];
    const float* frag_raw = (const float*)d_in[1];
    const float* mol_W1 = (const float*)d_in[2];  const float* mol_b1 = (const float*)d_in[3];
    const float* mol_g  = (const float*)d_in[4];  const float* mol_be = (const float*)d_in[5];
    const float* mol_W2 = (const float*)d_in[6];  const float* mol_b2 = (const float*)d_in[7];
    const float* frag_W1= (const float*)d_in[8];  const float* frag_b1= (const float*)d_in[9];
    const float* frag_g = (const float*)d_in[10]; const float* frag_be= (const float*)d_in[11];
    const float* frag_W2= (const float*)d_in[12]; const float* frag_b2= (const float*)d_in[13];
    const float* view_W1= (const float*)d_in[14]; const float* view_b1= (const float*)d_in[15];
    const float* view_g = (const float*)d_in[16]; const float* view_be= (const float*)d_in[17];
    const float* view_W2= (const float*)d_in[18]; const float* view_b2= (const float*)d_in[19];
    const float* enc_in_W = (const float*)d_in[20]; const float* enc_in_b = (const float*)d_in[21];
    const float* enc_out_W= (const float*)d_in[22]; const float* enc_out_b= (const float*)d_in[23];
    const float* Wq = (const float*)d_in[24]; const float* bq = (const float*)d_in[25];
    const float* Wk = (const float*)d_in[26]; const float* bk = (const float*)d_in[27];
    const float* Wv = (const float*)d_in[28]; const float* bv = (const float*)d_in[29];
    const float* Wo = (const float*)d_in[30]; const float* bo = (const float*)d_in[31];
    const float* Wf1= (const float*)d_in[32]; const float* bf1= (const float*)d_in[33];
    const float* Wf2= (const float*)d_in[34]; const float* bf2= (const float*)d_in[35];
    const float* ln1_g = (const float*)d_in[36]; const float* ln1_b = (const float*)d_in[37];
    const float* ln2_g = (const float*)d_in[38]; const float* ln2_b = (const float*)d_in[39];
    const int* sbn = (const int*)d_in[40];
    const int* pm  = (const int*)d_in[41];
    const int* pp  = (const int*)d_in[42];

    float* out      = (float*)d_out;
    float* mol_out  = out;                          // [512,256]
    float* view_out = out + 131072;                 // [512,256]
    float* sim_out  = out + 262144;                 // [6336,6336]

    float* ws = (float*)d_ws;
    size_t o = 0;
    auto alloc = [&](size_t n) { float* p = ws + o; o += n; return p; };
    float* pairs   = alloc((size_t)P_PAIRS * F_DIM);
    float* hfrag   = alloc((size_t)P_PAIRS * PH_DIM);
    float* fpre    = alloc((size_t)P_PAIRS * F_DIM);
    float* fproj   = alloc((size_t)P_PAIRS * F_DIM);
    float* headh   = alloc((size_t)B_MOL * PH_DIM);      // reused by mol & view heads
    float* headpre = alloc((size_t)B_MOL * F_DIM);
    float* batched = alloc((size_t)B_MOL * MAXP * F_DIM);
    float* hstate  = alloc((size_t)B_MOL * MAXP * H_DIM);
    float* lnout   = alloc((size_t)B_MOL * MAXP * H_DIM); // reused as enc output
    float* qb      = alloc((size_t)B_MOL * MAXP * H_DIM);
    float* kb      = alloc((size_t)B_MOL * MAXP * H_DIM);
    float* vb      = alloc((size_t)B_MOL * MAXP * H_DIM);
    float* att     = alloc((size_t)B_MOL * MAXP * H_DIM);
    float* ffn     = alloc((size_t)B_MOL * MAXP * FFN_DIM);
    float* rdout   = alloc((size_t)B_MOL * F_DIM);
    float* mu      = alloc(1024);
    float* inv     = alloc(1024);

    const int R = B_MOL * MAXP;   // 12288 encoder rows

    // ---- fragment pairs + frag projection head + similarity matrix
    pairs_sum<<<P_PAIRS, 256, 0, stream>>>(frag_raw, pairs);
    gemm_wmma_bf16<<<gemm_grid(P_PAIRS, PH_DIM), 256, 0, stream>>>(
        pairs, frag_W1, hfrag, frag_b1, nullptr, P_PAIRS, PH_DIM, F_DIM, 0, 0);
    bn_stats<<<PH_DIM / 32, 256, 0, stream>>>(hfrag, mu, inv, P_PAIRS, PH_DIM);
    bn_apply_relu<<<(P_PAIRS * PH_DIM + 255) / 256, 256, 0, stream>>>(
        hfrag, mu, inv, frag_g, frag_be, P_PAIRS, PH_DIM);
    gemm_wmma_bf16<<<gemm_grid(P_PAIRS, F_DIM), 256, 0, stream>>>(
        hfrag, frag_W2, fpre, frag_b2, nullptr, P_PAIRS, F_DIM, PH_DIM, 0, 0);
    l2norm_rows<<<P_PAIRS, 256, 0, stream>>>(fpre, fproj);
    gemm_wmma_bf16<<<gemm_grid(P_PAIRS, P_PAIRS), 256, 0, stream>>>(
        fproj, fproj, sim_out, nullptr, nullptr, P_PAIRS, P_PAIRS, F_DIM, 1, 0);

    // ---- mol projection head
    gemm_wmma_bf16<<<gemm_grid(B_MOL, PH_DIM), 256, 0, stream>>>(
        mol_emb, mol_W1, headh, mol_b1, nullptr, B_MOL, PH_DIM, F_DIM, 0, 0);
    bn_stats<<<PH_DIM / 32, 256, 0, stream>>>(headh, mu, inv, B_MOL, PH_DIM);
    bn_apply_relu<<<(B_MOL * PH_DIM + 255) / 256, 256, 0, stream>>>(
        headh, mu, inv, mol_g, mol_be, B_MOL, PH_DIM);
    gemm_wmma_bf16<<<gemm_grid(B_MOL, F_DIM), 256, 0, stream>>>(
        headh, mol_W2, headpre, mol_b2, nullptr, B_MOL, F_DIM, PH_DIM, 0, 0);
    l2norm_rows<<<B_MOL, 256, 0, stream>>>(headpre, mol_out);

    // ---- ragged pack + encoder input
    zero_f32<<<(R * F_DIM + 255) / 256, 256, 0, stream>>>(batched, (size_t)R * F_DIM);
    scatter_pairs<<<P_PAIRS, 256, 0, stream>>>(pairs, pm, pp, batched);
    gemm_wmma_bf16<<<gemm_grid(R, H_DIM), 256, 0, stream>>>(
        batched, enc_in_W, hstate, enc_in_b, nullptr, R, H_DIM, F_DIM, 0, 0);

    // ---- 2 pre-LN encoder layers
    for (int l = 0; l < 2; ++l) {
        layernorm_rows<<<R, 256, 0, stream>>>(hstate, lnout, ln1_g + l * 256, ln1_b + l * 256);
        gemm_wmma_bf16<<<gemm_grid(R, H_DIM), 256, 0, stream>>>(
            lnout, Wq + l * 65536, qb, bq + l * 256, nullptr, R, H_DIM, H_DIM, 0, 0);
        gemm_wmma_bf16<<<gemm_grid(R, H_DIM), 256, 0, stream>>>(
            lnout, Wk + l * 65536, kb, bk + l * 256, nullptr, R, H_DIM, H_DIM, 0, 0);
        gemm_wmma_bf16<<<gemm_grid(R, H_DIM), 256, 0, stream>>>(
            lnout, Wv + l * 65536, vb, bv + l * 256, nullptr, R, H_DIM, H_DIM, 0, 0);
        attention<<<B_MOL * 8, 256, 0, stream>>>(qb, kb, vb, att);
        gemm_wmma_bf16<<<gemm_grid(R, H_DIM), 256, 0, stream>>>(
            att, Wo + l * 65536, hstate, bo + l * 256, hstate, R, H_DIM, H_DIM, 0, 0);
        layernorm_rows<<<R, 256, 0, stream>>>(hstate, lnout, ln2_g + l * 256, ln2_b + l * 256);
        gemm_wmma_bf16<<<gemm_grid(R, FFN_DIM), 256, 0, stream>>>(
            lnout, Wf1 + l * 262144, ffn, bf1 + l * 1024, nullptr, R, FFN_DIM, H_DIM, 0, 2);
        gemm_wmma_bf16<<<gemm_grid(R, H_DIM), 256, 0, stream>>>(
            ffn, Wf2 + l * 262144, hstate, bf2 + l * 256, hstate, R, H_DIM, FFN_DIM, 0, 0);
    }

    // ---- encoder output, masked-mean readout, view projection head
    gemm_wmma_bf16<<<gemm_grid(R, F_DIM), 256, 0, stream>>>(
        hstate, enc_out_W, lnout, enc_out_b, nullptr, R, F_DIM, H_DIM, 0, 0);
    readout_mean<<<B_MOL, 256, 0, stream>>>(lnout, sbn, rdout);
    gemm_wmma_bf16<<<gemm_grid(B_MOL, PH_DIM), 256, 0, stream>>>(
        rdout, view_W1, headh, view_b1, nullptr, B_MOL, PH_DIM, F_DIM, 0, 0);
    bn_stats<<<PH_DIM / 32, 256, 0, stream>>>(headh, mu, inv, B_MOL, PH_DIM);
    bn_apply_relu<<<(B_MOL * PH_DIM + 255) / 256, 256, 0, stream>>>(
        headh, mu, inv, view_g, view_be, B_MOL, PH_DIM);
    gemm_wmma_bf16<<<gemm_grid(B_MOL, F_DIM), 256, 0, stream>>>(
        headh, view_W2, headpre, view_b2, nullptr, B_MOL, F_DIM, PH_DIM, 0, 0);
    l2norm_rows<<<B_MOL, 256, 0, stream>>>(headpre, view_out);
}
